// ArbitraryStructureRNN_26817775796427
// MI455X (gfx1250) — matrix operand order repaired and checked by reading, not compiled, not run on hardware
//
#include <hip/hip_runtime.h>
#include <hip/hip_bf16.h>

// ---------------------------------------------------------------------------
// ArbitraryStructureRNN on MI455X (gfx1250): bf16 WMMA GEMMs everywhere.
//   proj:  pre = X @ (W*mask)^T + (b_ih + b_hh)   (time-batched, parallel)
//          -> stored FRAGMENT-MAJOR (WMMA D-layout, 32B/lane) for the scan
//   scan:  h'  = tanh(pre[t] + h @ W_hh^T), sharded over 8 WGs:
//          each WG owns 64 columns of H, keeps its W_hh slice resident in LDS
//          for all 512 steps, and exchanges h via L2 with a split-barrier.
// ---------------------------------------------------------------------------

typedef __attribute__((ext_vector_type(16))) __bf16 v16bf;
typedef __attribute__((ext_vector_type(8)))  __bf16 v8bf;
typedef __attribute__((ext_vector_type(8)))  float  v8f;

static constexpr int Tt = 512;     // seq len
static constexpr int Bb = 64;      // batch
static constexpr int IN = 256;     // input size
static constexpr int Hh = 512;     // hidden size
static constexpr int NT = Hh / 16; // 32 N-tiles
static constexpr int NWG = 8;      // scan workgroups (each owns Hh/NWG = 64 cols)

union AFrag { v16bf v; v8bf h[2]; };

// fragment-major pre layout: tile (mtileG, ntile) -> 256 floats, lane*8 + v
__device__ __forceinline__ size_t frag_off(int mtileG, int ntile, int lane) {
  return ((size_t)mtileG * NT + ntile) * 256 + (size_t)lane * 8;
}

__device__ __forceinline__ float fast_tanh(float x) {
#if __has_builtin(__builtin_amdgcn_tanhf)
  return __builtin_amdgcn_tanhf(x);     // native V_TANH_F32 transcendental
#else
  return tanhf(x);
#endif
}

// ---------------------------------------------------------------------------
// weight packing: bf16(W * mask)   (mask may be null)
// ---------------------------------------------------------------------------
__global__ void pack_w_kernel(const float* __restrict__ W,
                              const float* __restrict__ mask,
                              __bf16* __restrict__ out, int n) {
  int i = blockIdx.x * blockDim.x + threadIdx.x;
  if (i < n) {
    float v = W[i];
    if (mask) v *= mask[i];
    out[i] = (__bf16)v;
  }
}

__global__ void bias_sum_kernel(const float* __restrict__ a,
                                const float* __restrict__ b,
                                float* __restrict__ out, int n) {
  int i = blockIdx.x * blockDim.x + threadIdx.x;
  if (i < n) out[i] = a[i] + b[i];
}

// ---------------------------------------------------------------------------
// proj: PreFrag[tile] = X @ Wb^T + bias   (one 16x16 tile per wave)
//   block = 256 threads (8 waves); wave w -> M-tile (blockIdx.x*8 + w),
//   N-tile blockIdx.y.  A: 16x32 bf16 frag, B: 32x16 bf16 frag (contig K).
// ---------------------------------------------------------------------------
__global__ __launch_bounds__(256)
void proj_kernel(const void* __restrict__ Xv, int x_is_f32,
                 const __bf16* __restrict__ Wb,
                 const float* __restrict__ bias,
                 float* __restrict__ preF, int K) {
  const int lane  = threadIdx.x & 31;
  const int wave  = threadIdx.x >> 5;
  const int mtile = blockIdx.x * 8 + wave;
  const int ntile = blockIdx.y;
  const int hi    = lane >> 4;
  const int lc    = lane & 15;
  const int arow  = mtile * 16 + lc;  // A: lane holds one M row
  const int khalf = hi * 8;           // A half-lane K split
  const int bn    = ntile * 16 + lc;  // B: lane holds one N column
  const int bko   = hi * 8;           // B half-lane K split

  v8f acc = {};
  const __bf16* Wn = Wb + (size_t)bn * K;

  if (x_is_f32) {
    const float* row = (const float*)Xv + (size_t)arow * K;
    for (int kc = 0; kc < K; kc += 32) {
      v16bf a;
#pragma unroll
      for (int i = 0; i < 8; ++i) {
        a[i]     = (__bf16)row[kc + khalf + i];
        a[8 + i] = (__bf16)row[kc + khalf + 16 + i];
      }
      AFrag b;
      b.h[0] = *(const v8bf*)(Wn + kc + bko);
      b.h[1] = *(const v8bf*)(Wn + kc + bko + 16);
      acc = __builtin_amdgcn_wmma_f32_16x16x32_bf16(false, a, false, b.v,
                                                    (short)0, acc, false, false);
    }
  } else {
    const __bf16* row = (const __bf16*)Xv + (size_t)arow * K;
    for (int kc = 0; kc < K; kc += 32) {
      AFrag a, b;
      a.h[0] = *(const v8bf*)(row + kc + khalf);
      a.h[1] = *(const v8bf*)(row + kc + khalf + 16);
      b.h[0] = *(const v8bf*)(Wn + kc + bko);
      b.h[1] = *(const v8bf*)(Wn + kc + bko + 16);
      acc = __builtin_amdgcn_wmma_f32_16x16x32_bf16(false, a.v, false, b.v,
                                                    (short)0, acc, false, false);
    }
  }

  const float bv = bias[bn];
#pragma unroll
  for (int v = 0; v < 8; ++v) acc[v] += bv;

  *(v8f*)(preF + frag_off(mtile, ntile, lane)) = acc;   // 2x global_store_b128
}

// ---------------------------------------------------------------------------
// scan: 8 WGs x 256 threads.  WG g owns N columns [g*64, g*64+64).
//   LDS: hbuf (64KB, full 64x512 h in bf16) + wlds (64KB, W_hh slice,
//   resident across all timesteps).  Per step: 256 WMMAs per WG from LDS
//   operands, C from fragment-major pre, h exchanged via L2 + split barrier.
// ---------------------------------------------------------------------------
__global__ __launch_bounds__(256)
void scan_kernel(const float* __restrict__ preF,
                 const __bf16* __restrict__ Whh,
                 __bf16* __restrict__ hg,              // bf16 [2][Bb][Hh] ping-pong
                 unsigned long long* __restrict__ cnt, // arrival counter (pre-zeroed)
                 __bf16* __restrict__ outseq,          // bf16 [T,B,H] or null
                 float* __restrict__ final_out,        // f32 [B,H] or null
                 int add_final) {
  extern __shared__ char smem[];
  __bf16* hbuf = (__bf16*)smem;                 // 64 KB: full h_t
  __bf16* wlds = (__bf16*)(smem + Bb * Hh * 2); // 64 KB: W slice [64 cols][512 k]

  const int tid   = threadIdx.x;
  const int lane  = tid & 31;
  const int wave  = tid >> 5;          // 0..7
  const int g     = blockIdx.x;        // 0..7
  const int ntl   = wave & 3;          // local N-tile 0..3
  const int mhalf = wave >> 2;         // 0..1 -> M-tiles {2*mhalf, 2*mhalf+1}
  const int ntile = g * 4 + ntl;       // global N-tile
  const int hi    = lane >> 4;
  const int lc    = lane & 15;
  const int khalf = hi * 8;

  // ---- preload W_hh slice (contiguous 64 rows of [Hh]) into LDS ----
  {
    const v8bf* src = (const v8bf*)(Whh + (size_t)g * 64 * Hh);
    v8bf*       dst = (v8bf*)wlds;
    for (int i = tid; i < (64 * Hh) / 8; i += 256) dst[i] = src[i];
  }
  // ---- h0 = 0 ----
  for (int i = tid; i < Bb * Hh; i += 256) hbuf[i] = (__bf16)0.0f;
  __syncthreads();

  const int arow0 = (mhalf * 2 + 0) * 16 + lc;
  const int arow1 = (mhalf * 2 + 1) * 16 + lc;
  const __bf16* wp = wlds + (size_t)(ntl * 16 + lc) * Hh + hi * 8;

  for (int t = 0; t < Tt; ++t) {
    // C init: fragment-major pre (2x global_load_b128 per tile)
    v8f acc0 = *(const v8f*)(preF + frag_off(t * 4 + mhalf * 2 + 0, ntile, lane));
    v8f acc1 = *(const v8f*)(preF + frag_off(t * 4 + mhalf * 2 + 1, ntile, lane));

    for (int kc = 0; kc < Hh; kc += 32) {
      AFrag b;                                  // B from LDS-resident W slice
      b.h[0] = *(const v8bf*)(wp + kc);
      b.h[1] = *(const v8bf*)(wp + kc + 16);
      AFrag a0, a1;                             // A from LDS h
      a0.h[0] = *(const v8bf*)(&hbuf[arow0 * Hh + kc + khalf]);
      a0.h[1] = *(const v8bf*)(&hbuf[arow0 * Hh + kc + khalf + 16]);
      a1.h[0] = *(const v8bf*)(&hbuf[arow1 * Hh + kc + khalf]);
      a1.h[1] = *(const v8bf*)(&hbuf[arow1 * Hh + kc + khalf + 16]);
      acc0 = __builtin_amdgcn_wmma_f32_16x16x32_bf16(false, a0.v, false, b.v,
                                                     (short)0, acc0, false, false);
      acc1 = __builtin_amdgcn_wmma_f32_16x16x32_bf16(false, a1.v, false, b.v,
                                                     (short)0, acc1, false, false);
    }

    // ---- activation + publish this WG's 64-column slice of h_{t+1} ----
    __bf16* hcur = hg + (size_t)(t & 1) * Bb * Hh;
    const int col = ntile * 16 + lc;
#pragma unroll
    for (int i = 0; i < 2; ++i) {
      v8f& acc = i ? acc1 : acc0;
      const int mrow = (mhalf * 2 + i) * 16 + hi * 8;
#pragma unroll
      for (int v = 0; v < 8; ++v) {
        const int row = mrow + v;
        const float hval = fast_tanh(acc[v]);
        const __bf16 hb = (__bf16)hval;
        hcur[row * Hh + col] = hb;
        if (outseq)
          outseq[(size_t)t * Bb * Hh + (size_t)row * Hh + col] = hb;
        if (final_out && t == Tt - 1) {
          const size_t o = (size_t)row * Hh + col;
          final_out[o] = add_final ? (final_out[o] + hval) : hval;
        }
      }
    }

    if (t == Tt - 1) break;   // no exchange needed after the last step

    // ---- split barrier across the 8 WGs (release -> spin acquire) ----
    __threadfence();                      // make h slice visible device-wide
    __syncthreads();
#if __has_builtin(__builtin_amdgcn_s_cluster_barrier)
    __builtin_amdgcn_s_cluster_barrier(); // HW cluster barrier (S_NOP if unclustered)
#endif
    if (tid == 0) {
      __hip_atomic_fetch_add(cnt, 1ull, __ATOMIC_RELEASE, __HIP_MEMORY_SCOPE_AGENT);
      const unsigned long long target = (unsigned long long)NWG * (t + 1);
      while (__hip_atomic_load(cnt, __ATOMIC_ACQUIRE, __HIP_MEMORY_SCOPE_AGENT) < target)
        __builtin_amdgcn_s_sleep(2);
    }
    __syncthreads();

    // ---- pull the full h_{t+1} (all columns) into LDS for the next step ----
    {
      const v8bf* src = (const v8bf*)hcur;
      v8bf*       dst = (v8bf*)hbuf;
      for (int i = tid; i < (Bb * Hh) / 8; i += 256) dst[i] = src[i];
    }
    __syncthreads();
  }
}

// ---------------------------------------------------------------------------
// driver
// ---------------------------------------------------------------------------
extern "C" void kernel_launch(void* const* d_in, const int* in_sizes, int n_in,
                              void* d_out, int out_size, void* d_ws, size_t ws_size,
                              hipStream_t stream) {
  const float* x     = (const float*)d_in[0];
  const float* W_ih0 = (const float*)d_in[1];
  const float* b_ih0 = (const float*)d_in[2];
  const float* W_hh0 = (const float*)d_in[3];
  const float* b_hh0 = (const float*)d_in[4];
  const float* W_ih1 = (const float*)d_in[5];
  const float* mask1 = (const float*)d_in[6];
  const float* b_ih1 = (const float*)d_in[7];
  const float* W_hh1 = (const float*)d_in[8];
  const float* b_hh1 = (const float*)d_in[9];
  const float* W_ih2 = (const float*)d_in[10];
  const float* mask2 = (const float*)d_in[11];
  const float* b_ih2 = (const float*)d_in[12];
  const float* W_hh2 = (const float*)d_in[13];
  const float* b_hh2 = (const float*)d_in[14];
  const float* W_ihs = (const float*)d_in[15];
  const float* masks = (const float*)d_in[16];
  const float* b_ihs = (const float*)d_in[17];
  const float* W_hhs = (const float*)d_in[18];
  const float* b_hhs = (const float*)d_in[19];

  char*  ws  = (char*)d_ws;
  size_t off = 0;
  auto alloc = [&](size_t bytes) -> void* {
    void* p = ws + off;
    off += (bytes + 255) & ~(size_t)255;
    return p;
  };

  const int M = Tt * Bb;                              // 32768 rows
  __bf16* wb_ih0 = (__bf16*)alloc((size_t)Hh * IN * 2);
  __bf16* wb_hh0 = (__bf16*)alloc((size_t)Hh * Hh * 2);
  __bf16* wb_ih1 = (__bf16*)alloc((size_t)Hh * Hh * 2);
  __bf16* wb_hh1 = (__bf16*)alloc((size_t)Hh * Hh * 2);
  __bf16* wb_ih2 = (__bf16*)alloc((size_t)Hh * Hh * 2);
  __bf16* wb_hh2 = (__bf16*)alloc((size_t)Hh * Hh * 2);
  __bf16* wb_ihs = (__bf16*)alloc((size_t)Hh * Hh * 2);
  __bf16* wb_hhs = (__bf16*)alloc((size_t)Hh * Hh * 2);
  float*  bias0  = (float*)alloc((size_t)Hh * 4);
  float*  bias1  = (float*)alloc((size_t)Hh * 4);
  float*  bias2  = (float*)alloc((size_t)Hh * 4);
  float*  biass  = (float*)alloc((size_t)Hh * 4);
  float*  preF   = (float*)alloc((size_t)M * Hh * 4);   // 64 MB, fragment-major
  __bf16* out0   = (__bf16*)alloc((size_t)M * Hh * 2);  // 32 MB, row-major bf16
  __bf16* out1   = (__bf16*)alloc((size_t)M * Hh * 2);  // 32 MB
  __bf16* hg     = (__bf16*)alloc((size_t)2 * Bb * Hh * 2);  // h ping-pong
  unsigned long long* cnt = (unsigned long long*)alloc(8);

  // ---- pack weights (fuse masks) + bias sums ----
  const int nw_in = Hh * IN, nw_hh = Hh * Hh;
  dim3 pb(256);
  pack_w_kernel<<<dim3((nw_in + 255) / 256), pb, 0, stream>>>(W_ih0, nullptr, wb_ih0, nw_in);
  pack_w_kernel<<<dim3((nw_hh + 255) / 256), pb, 0, stream>>>(W_hh0, nullptr, wb_hh0, nw_hh);
  pack_w_kernel<<<dim3((nw_hh + 255) / 256), pb, 0, stream>>>(W_ih1, mask1,  wb_ih1, nw_hh);
  pack_w_kernel<<<dim3((nw_hh + 255) / 256), pb, 0, stream>>>(W_hh1, nullptr, wb_hh1, nw_hh);
  pack_w_kernel<<<dim3((nw_hh + 255) / 256), pb, 0, stream>>>(W_ih2, mask2,  wb_ih2, nw_hh);
  pack_w_kernel<<<dim3((nw_hh + 255) / 256), pb, 0, stream>>>(W_hh2, nullptr, wb_hh2, nw_hh);
  pack_w_kernel<<<dim3((nw_hh + 255) / 256), pb, 0, stream>>>(W_ihs, masks,  wb_ihs, nw_hh);
  pack_w_kernel<<<dim3((nw_hh + 255) / 256), pb, 0, stream>>>(W_hhs, nullptr, wb_hhs, nw_hh);
  bias_sum_kernel<<<dim3((Hh + 255) / 256), pb, 0, stream>>>(b_ih0, b_hh0, bias0, Hh);
  bias_sum_kernel<<<dim3((Hh + 255) / 256), pb, 0, stream>>>(b_ih1, b_hh1, bias1, Hh);
  bias_sum_kernel<<<dim3((Hh + 255) / 256), pb, 0, stream>>>(b_ih2, b_hh2, bias2, Hh);
  bias_sum_kernel<<<dim3((Hh + 255) / 256), pb, 0, stream>>>(b_ihs, b_hhs, biass, Hh);

  const dim3 pgrid(M / 16 / 8, Hh / 16);   // 8 M-tiles per block, 1 N-tile
  const dim3 pblock(256);
  const size_t scan_lds = (size_t)(Bb * Hh + 64 * Hh) * 2;   // 128 KB of 320 KB

  // ---- layer 0 ----
  proj_kernel<<<pgrid, pblock, 0, stream>>>(x, 1, wb_ih0, bias0, preF, IN);
  hipMemsetAsync(cnt, 0, 8, stream);
  scan_kernel<<<NWG, 256, scan_lds, stream>>>(preF, wb_hh0, hg, cnt, out0, nullptr, 0);
  // ---- layer 1 ----
  proj_kernel<<<pgrid, pblock, 0, stream>>>(out0, 0, wb_ih1, bias1, preF, Hh);
  hipMemsetAsync(cnt, 0, 8, stream);
  scan_kernel<<<NWG, 256, scan_lds, stream>>>(preF, wb_hh1, hg, cnt, out1, nullptr, 0);
  // ---- layer 2 (final state -> d_out) ----
  proj_kernel<<<pgrid, pblock, 0, stream>>>(out1, 0, wb_ih2, bias2, preF, Hh);
  hipMemsetAsync(cnt, 0, 8, stream);
  scan_kernel<<<NWG, 256, scan_lds, stream>>>(preF, wb_hh2, hg, cnt, nullptr, (float*)d_out, 0);
  // ---- skip RNN from layer-0 output (final state += d_out) ----
  proj_kernel<<<pgrid, pblock, 0, stream>>>(out0, 0, wb_ihs, biass, preF, Hh);
  hipMemsetAsync(cnt, 0, 8, stream);
  scan_kernel<<<NWG, 256, scan_lds, stream>>>(preF, wb_hhs, hg, cnt, nullptr, (float*)d_out, 1);
}
